// BitNetAttention_29265907155294
// MI455X (gfx1250) — 16-bit and 8-bit matrix operand order repaired and checked by reading, compiled, not run on hardware
//
#include <hip/hip_runtime.h>

#define HIDDEN 2048
#define NHEADS 16
#define HDIM   128
#define SEQ    2048
#define BATCH  2
#define QBF    127.0f

typedef __attribute__((ext_vector_type(8)))  int    v8i;
typedef __attribute__((ext_vector_type(8)))  float  v8f;
typedef __attribute__((ext_vector_type(16))) __bf16 v16bf;

static __device__ __forceinline__ float redmax16(float v) {
#pragma unroll
  for (int m = 1; m < 16; m <<= 1) v = fmaxf(v, __shfl_xor(v, m, 32));
  return v;
}
static __device__ __forceinline__ float redsum16(float v) {
#pragma unroll
  for (int m = 1; m < 16; m <<= 1) v += __shfl_xor(v, m, 32);
  return v;
}

static __device__ __forceinline__ void async_copy_b128(unsigned lds_off,
                                                       unsigned long long gaddr) {
  asm volatile("global_load_async_to_lds_b128 %0, %1, off"
               :: "v"(lds_off), "v"(gaddr) : "memory");
}
static __device__ __forceinline__ void wait_async0() {
  asm volatile("s_wait_asynccnt 0x0" ::: "memory");
}

__global__ void k_zero(float* p, int n) {
  int i = blockIdx.x * blockDim.x + threadIdx.x;
  if (i < n) p[i] = 0.0f;
}

__global__ void k_abssum(const float* __restrict__ w, float* __restrict__ sum, int n) {
  __shared__ float s[256];
  float acc = 0.0f;
  for (int i = blockIdx.x * 256 + threadIdx.x; i < n; i += gridDim.x * 256)
    acc += fabsf(w[i]);
  s[threadIdx.x] = acc;
  __syncthreads();
  for (int st = 128; st > 0; st >>= 1) {
    if (threadIdx.x < st) s[threadIdx.x] += s[threadIdx.x + st];
    __syncthreads();
  }
  if (threadIdx.x == 0) atomicAdd(sum, s[0]);
}

__global__ void k_wquant(const float* __restrict__ w, const float* __restrict__ sum,
                         signed char* __restrict__ q, int n) {
  float mean  = fmaxf(sum[0] / (float)n, 1e-5f);
  float scale = 1.0f / mean;
  int i = blockIdx.x * 256 + threadIdx.x;
  if (i < n) {
    float v = rintf(w[i] * scale);
    q[i] = (signed char)fminf(fmaxf(v, -1.0f), 1.0f);
  }
}

__global__ __launch_bounds__(256) void k_actquant(const float* __restrict__ x,
                                                  signed char* __restrict__ q,
                                                  float* __restrict__ deq, int ncols) {
  __shared__ float s[256];
  int row = blockIdx.x;
  const float* xr = x + (size_t)row * ncols;
  float mx = 0.0f;
  for (int c = threadIdx.x; c < ncols; c += 256) mx = fmaxf(mx, fabsf(xr[c]));
  s[threadIdx.x] = mx;
  __syncthreads();
  for (int st = 128; st > 0; st >>= 1) {
    if (threadIdx.x < st) s[threadIdx.x] = fmaxf(s[threadIdx.x], s[threadIdx.x + st]);
    __syncthreads();
  }
  float m = fmaxf(s[0], 1e-5f);
  float scale = QBF / m;
  if (threadIdx.x == 0) deq[row] = m / QBF;
  signed char* qr = q + (size_t)row * ncols;
  for (int c = threadIdx.x; c < ncols; c += 256) {
    float v = rintf(xr[c] * scale);
    qr[c] = (signed char)fminf(fmaxf(v, -QBF), QBF);
  }
}

__global__ __launch_bounds__(256) void k_gemm_iu8(const signed char* __restrict__ X,
                                                  const signed char* __restrict__ W,
                                                  const float* __restrict__ xdeq,
                                                  const float* __restrict__ wsum,
                                                  float* __restrict__ Y,
                                                  int M, int N, int K, float wcount) {
  const int lane = threadIdx.x & 31;
  const int wave = threadIdx.x >> 5;
  const int wm = wave >> 1, wn = wave & 1;
  const int m0 = blockIdx.x * 128 + wm * 32;
  const int n0 = blockIdx.y * 128 + wn * 64;
  const int half = lane >> 4;
  const int l16  = lane & 15;
  const int kbA = half * 16;
  const int kbB = half * 16;

  v8i acc[2][4];
#pragma unroll
  for (int i = 0; i < 2; i++)
#pragma unroll
    for (int j = 0; j < 4; j++) acc[i][j] = (v8i){};

  for (int k0 = 0; k0 < K; k0 += 64) {
    v8i a[2], b[4];
#pragma unroll
    for (int mi = 0; mi < 2; mi++) {
      const signed char* base = X + (size_t)(m0 + mi * 16 + l16) * K + k0 + kbA;
      int2 t0 = *(const int2*)(base + 0);
      int2 t1 = *(const int2*)(base + 8);
      int2 t2 = *(const int2*)(base + 32);
      int2 t3 = *(const int2*)(base + 40);
      v8i av;
      av[0] = t0.x; av[1] = t0.y; av[2] = t1.x; av[3] = t1.y;
      av[4] = t2.x; av[5] = t2.y; av[6] = t3.x; av[7] = t3.y;
      a[mi] = av;
    }
#pragma unroll
    for (int ni = 0; ni < 4; ni++) {
      const signed char* base = W + (size_t)(n0 + ni * 16 + l16) * K + k0 + kbB;
      int4 t0 = *(const int4*)(base);
      int4 t1 = *(const int4*)(base + 32);
      v8i bv;
      bv[0] = t0.x; bv[1] = t0.y; bv[2] = t0.z; bv[3] = t0.w;
      bv[4] = t1.x; bv[5] = t1.y; bv[6] = t1.z; bv[7] = t1.w;
      b[ni] = bv;
    }
#pragma unroll
    for (int mi = 0; mi < 2; mi++)
#pragma unroll
      for (int ni = 0; ni < 4; ni++)
        acc[mi][ni] = __builtin_amdgcn_wmma_i32_16x16x64_iu8(
            true, a[mi], true, b[ni], acc[mi][ni], false, false);
  }

  float wdeq = fmaxf(wsum[0] / wcount, 1e-5f);
#pragma unroll
  for (int mi = 0; mi < 2; mi++)
#pragma unroll
    for (int ni = 0; ni < 4; ni++)
#pragma unroll
      for (int r = 0; r < 8; r++) {
        int m = m0 + mi * 16 + r + half * 8;
        int n = n0 + ni * 16 + l16;
        Y[(size_t)m * N + n] = (float)acc[mi][ni][r] * xdeq[m] * wdeq;
      }
}

__global__ void k_pack(const float* __restrict__ T, const int* __restrict__ posids,
                       __bf16* __restrict__ dst, int doRope, int doTrans, float qscale) {
  size_t i = (size_t)blockIdx.x * 256 + threadIdx.x;
  size_t total = (size_t)BATCH * SEQ * HIDDEN;
  if (i >= total) return;
  int col = (int)(i % HIDDEN);
  size_t row = i / HIDDEN;
  int b = (int)(row / SEQ), s = (int)(row % SEQ);
  int h = col / HDIM, d = col % HDIM;
  float v = T[i];
  if (doRope) {
    int pos = posids[(size_t)b * SEQ + s];
    int dh = d & 63;
    float fr = (float)pos * __powf(10000.0f, -(float)dh / 64.0f);
    float c = __cosf(fr), sn = __sinf(fr);
    float other = (d < 64) ? -T[row * HIDDEN + h * HDIM + d + 64]
                           :  T[row * HIDDEN + h * HDIM + d - 64];
    v = v * c + other * sn;
  }
  v *= qscale;
  size_t bh = (size_t)b * NHEADS + h;
  if (doTrans) dst[(bh * HDIM + d) * SEQ + s] = (__bf16)v;
  else         dst[(bh * SEQ + s) * HDIM + d] = (__bf16)v;
}

__global__ __launch_bounds__(256) void k_flash(const __bf16* __restrict__ Q,
                                               const __bf16* __restrict__ Kb,
                                               const __bf16* __restrict__ Vt,
                                               float* __restrict__ O) {
  __shared__ __align__(16) __bf16 plds[8][16][32];
  __shared__ __align__(16) __bf16 ksh[32][HDIM];
  __shared__ __align__(16) __bf16 vsh[HDIM][32];
  const int tid  = threadIdx.x;
  const int lane = tid & 31;
  const int wave = tid >> 5;
  const int half = lane >> 4, l16 = lane & 15;
  const int bh = blockIdx.y;
  const int b = bh >> 4, h = bh & 15;
  const int q0 = blockIdx.x * 128 + wave * 16;
  const __bf16* Qp = Q  + (size_t)bh * SEQ * HDIM;
  const __bf16* Kp = Kb + (size_t)bh * SEQ * HDIM;
  const __bf16* Vp = Vt + (size_t)bh * HDIM * SEQ;

  v16bf qa[4];
  {
    const __bf16* qrow = Qp + (size_t)(q0 + l16) * HDIM;
#pragma unroll
    for (int c = 0; c < 4; c++) {
      union { v16bf v; int4 i4[2]; } u;
      u.i4[0] = *(const int4*)(qrow + c * 32 + half * 8);
      u.i4[1] = *(const int4*)(qrow + c * 32 + 16 + half * 8);
      qa[c] = u.v;
    }
  }
  v8f o[8];
  float mrow[8], lrow[8];
#pragma unroll
  for (int c = 0; c < 8; c++) o[c] = (v8f){};
#pragma unroll
  for (int r = 0; r < 8; r++) { mrow[r] = -1e30f; lrow[r] = 0.0f; }

  const unsigned kshOff = (unsigned)(size_t)((char*)&ksh[0][0] + tid * 32);
  const unsigned vshOff = (unsigned)(size_t)((char*)&vsh[0][0] + tid * 32);

  const int kend = min(SEQ, blockIdx.x * 128 + 128);
  for (int k0 = 0; k0 < kend; k0 += 32) {
    {
      const char* gk = (const char*)(Kp + (size_t)(k0 + (tid >> 3)) * HDIM) + (tid & 7) * 32;
      const char* gv = (const char*)(Vp + (size_t)(tid >> 1) * SEQ + k0) + (tid & 1) * 32;
      async_copy_b128(kshOff,      (unsigned long long)gk);
      async_copy_b128(kshOff + 16, (unsigned long long)(gk + 16));
      async_copy_b128(vshOff,      (unsigned long long)gv);
      async_copy_b128(vshOff + 16, (unsigned long long)(gv + 16));
      wait_async0();
      __syncthreads();
    }
    v8f sc[2];
#pragma unroll
    for (int t = 0; t < 2; t++) {
      v8f s = (v8f){};
#pragma unroll
      for (int c = 0; c < 4; c++) {
        union { v16bf v; int4 i4[2]; } u;
        const __bf16* krow = &ksh[t * 16 + l16][c * 32 + half * 8];
        u.i4[0] = *(const int4*)(krow);
        u.i4[1] = *(const int4*)(krow + 16);
        s = __builtin_amdgcn_wmma_f32_16x16x32_bf16(false, qa[c], false, u.v,
                                                    (short)0, s, false, false);
      }
      sc[t] = s;
    }
    float corr[8];
#pragma unroll
    for (int r = 0; r < 8; r++) {
      int qrow = q0 + r + half * 8;
      float s0 = sc[0][r] + ((k0 + l16)      > qrow ? -1e9f : 0.0f);
      float s1 = sc[1][r] + ((k0 + 16 + l16) > qrow ? -1e9f : 0.0f);
      float cm = redmax16(fmaxf(s0, s1));
      float mn = fmaxf(mrow[r], cm);
      float p0 = __expf(s0 - mn), p1 = __expf(s1 - mn);
      corr[r] = __expf(mrow[r] - mn);
      lrow[r] = lrow[r] * corr[r] + redsum16(p0 + p1);
      mrow[r] = mn;
      plds[wave][r + half * 8][l16]      = (__bf16)p0;
      plds[wave][r + half * 8][16 + l16] = (__bf16)p1;
    }
#pragma unroll
    for (int c = 0; c < 8; c++)
#pragma unroll
      for (int r = 0; r < 8; r++) o[c][r] *= corr[r];
    __syncthreads();
    union { v16bf v; int4 i4[2]; } pu;
    pu.i4[0] = *(const int4*)(&plds[wave][l16][half * 8]);
    pu.i4[1] = *(const int4*)(&plds[wave][l16][16 + half * 8]);
    v16bf pf = pu.v;
#pragma unroll
    for (int c = 0; c < 8; c++) {
      union { v16bf v; int4 i4[2]; } u;
      const __bf16* vrow = &vsh[c * 16 + l16][half * 8];
      u.i4[0] = *(const int4*)(vrow);
      u.i4[1] = *(const int4*)(vrow + 16);
      o[c] = __builtin_amdgcn_wmma_f32_16x16x32_bf16(false, pf, false, u.v,
                                                     (short)0, o[c], false, false);
    }
    __syncthreads();
  }
#pragma unroll
  for (int c = 0; c < 8; c++)
#pragma unroll
    for (int r = 0; r < 8; r++) {
      int qrow = q0 + r + half * 8;
      O[((size_t)b * SEQ + qrow) * HIDDEN + h * HDIM + c * 16 + l16] = o[c][r] / lrow[r];
    }
}

__global__ __launch_bounds__(256) void k_fwht_quant(const float* __restrict__ X,
                                                    signed char* __restrict__ q,
                                                    float* __restrict__ deq) {
  __shared__ float buf[HIDDEN];
  __shared__ float red[256];
  int row = blockIdx.x;
  const float* xr = X + (size_t)row * HIDDEN;
  for (int c = threadIdx.x; c < HIDDEN; c += 256) buf[c] = xr[c];
  __syncthreads();
  for (int hsz = 1; hsz < HIDDEN; hsz <<= 1) {
    for (int p = threadIdx.x; p < HIDDEN / 2; p += 256) {
      int pos = (p / hsz) * (hsz * 2) + (p % hsz);
      float a = buf[pos], bb = buf[pos + hsz];
      buf[pos] = a + bb;
      buf[pos + hsz] = a - bb;
    }
    __syncthreads();
  }
  const float norm = 0.022097086912079608f;
  float mx = 0.0f;
  for (int c = threadIdx.x; c < HIDDEN; c += 256) {
    float v = buf[c] * norm;
    buf[c] = v;
    mx = fmaxf(mx, fabsf(v));
  }
  red[threadIdx.x] = mx;
  __syncthreads();
  for (int st = 128; st > 0; st >>= 1) {
    if (threadIdx.x < st) red[threadIdx.x] = fmaxf(red[threadIdx.x], red[threadIdx.x + st]);
    __syncthreads();
  }
  float m = fmaxf(red[0], 1e-5f);
  float scale = QBF / m;
  if (threadIdx.x == 0) deq[row] = m / QBF;
  signed char* qr = q + (size_t)row * HIDDEN;
  for (int c = threadIdx.x; c < HIDDEN; c += 256) {
    float v = rintf(buf[c] * scale);
    qr[c] = (signed char)fminf(fmaxf(v, -QBF), QBF);
  }
}

extern "C" void kernel_launch(void* const* d_in, const int* in_sizes, int n_in,
                              void* d_out, int out_size, void* d_ws, size_t ws_size,
                              hipStream_t stream) {
  (void)in_sizes; (void)n_in; (void)out_size; (void)ws_size;
  const float* hs     = (const float*)d_in[0];
  const int*   posids = (const int*)d_in[2];
  const float* Wsrc[4] = {(const float*)d_in[3], (const float*)d_in[4],
                          (const float*)d_in[5], (const float*)d_in[6]};
  float* out = (float*)d_out;

  const int M = BATCH * SEQ;
  const size_t HH = (size_t)HIDDEN * HIDDEN;
  char* ws = (char*)d_ws;
  float*       wsum   = (float*)(ws + 0);
  float*       scaleA = (float*)(ws + 256);
  float*       scaleB = (float*)(ws + 256 + 16384);
  signed char* xq     = (signed char*)(ws + 33024);
  signed char* wqq    = (signed char*)(ws + 33024 + (size_t)M * HIDDEN);
  float*       temp   = (float*)((char*)wqq + 4 * HH);
  __bf16*      Qb     = (__bf16*)((char*)temp + (size_t)M * HIDDEN * 4);
  __bf16*      Kb     = Qb + (size_t)M * HIDDEN;
  __bf16*      Vt     = Kb + (size_t)M * HIDDEN;

  const float wcount = (float)HIDDEN * (float)HIDDEN;
  const int nElem = (int)HH;

  k_zero<<<1, 32, 0, stream>>>(wsum, 4);
  for (int i = 0; i < 4; i++)
    k_abssum<<<256, 256, 0, stream>>>(Wsrc[i], wsum + i, nElem);
  for (int i = 0; i < 4; i++)
    k_wquant<<<(nElem + 255) / 256, 256, 0, stream>>>(Wsrc[i], wsum + i,
                                                      wqq + (size_t)i * HH, nElem);
  k_actquant<<<M, 256, 0, stream>>>(hs, xq, scaleA, HIDDEN);

  dim3 gg(M / 128, HIDDEN / 128);
  int packBlocks = (int)(((size_t)M * HIDDEN) / 256);
  const float invSqrtD = 0.08838834764831845f;

  k_gemm_iu8<<<gg, 256, 0, stream>>>(xq, wqq + 0 * HH, scaleA, wsum + 0, temp,
                                     M, HIDDEN, HIDDEN, wcount);
  k_pack<<<packBlocks, 256, 0, stream>>>(temp, posids, Qb, 1, 0, invSqrtD);
  k_gemm_iu8<<<gg, 256, 0, stream>>>(xq, wqq + 1 * HH, scaleA, wsum + 1, temp,
                                     M, HIDDEN, HIDDEN, wcount);
  k_pack<<<packBlocks, 256, 0, stream>>>(temp, posids, Kb, 1, 0, 1.0f);
  k_gemm_iu8<<<gg, 256, 0, stream>>>(xq, wqq + 2 * HH, scaleA, wsum + 2, temp,
                                     M, HIDDEN, HIDDEN, wcount);
  k_pack<<<packBlocks, 256, 0, stream>>>(temp, posids, Vt, 0, 1, 1.0f);

  dim3 fg(SEQ / 128, BATCH * NHEADS);
  k_flash<<<fg, 256, 0, stream>>>(Qb, Kb, Vt, temp);

  k_fwht_quant<<<M, 256, 0, stream>>>(temp, xq, scaleB);
  k_gemm_iu8<<<gg, 256, 0, stream>>>(xq, wqq + 3 * HH, scaleB, wsum + 3, out,
                                     M, HIDDEN, HIDDEN, wcount);
}
